// AlphaEntmax_54631984005787
// MI455X (gfx1250) — matrix operand order repaired and checked by reading, not compiled, not run on hardware
//
#include <hip/hip_runtime.h>

typedef __attribute__((ext_vector_type(16))) __bf16 v16bf;
typedef __attribute__((ext_vector_type(8)))  float  v8f;

#define TOK  512   // B*C
#define EDIM 512
#define HDIM 8
#define EHD  4096  // E*H

// ---------------------------------------------------------------- f32 -> bf16
__global__ void f2bf_kernel(const float* __restrict__ in,
                            __bf16* __restrict__ out, int n) {
  int i = blockIdx.x * blockDim.x + threadIdx.x;
  if (i < n) out[i] = (__bf16)in[i];
}

// ---------------------------------------------------- bf16 WMMA GEMM + bias
// C[M,N] (f32) = A[M,K](bf16,row-major) @ B[K,N](bf16,row-major) + bias[N]
// Block: 256 thr (8 waves). Block tile 64x128, wave tile 32x32, K-step 32.
__global__ __launch_bounds__(256) void gemm_bias_bf16(
    const __bf16* __restrict__ A, const __bf16* __restrict__ B,
    const float* __restrict__ bias, float* __restrict__ C,
    int M, int N, int K) {
  __shared__ __align__(16) __bf16 As[64][32];
  __shared__ __align__(16) __bf16 Bst[128][32];  // transposed: [n][k]

  const int tid  = threadIdx.x;
  const int wave = tid >> 5;
  const int lane = tid & 31;
  const int l16  = lane & 15;
  const bool hi  = lane >= 16;
  const int wm = wave >> 2;            // 0..1
  const int wn = wave & 3;             // 0..3
  const int m0 = blockIdx.y * 64;
  const int n0 = blockIdx.x * 128;

  v8f acc[2][2];
#pragma unroll
  for (int mi = 0; mi < 2; ++mi)
#pragma unroll
    for (int ni = 0; ni < 2; ++ni)
#pragma unroll
      for (int r = 0; r < 8; ++r) acc[mi][ni][r] = 0.f;

  const int arow = tid >> 2;           // 0..63
  const int acol = (tid & 3) * 8;      // 0,8,16,24
  const int brow = tid >> 4;           // 0..15
  const int bcol = (tid & 15) * 8;     // 0..120

  for (int kk = 0; kk < K; kk += 32) {
    // stage A tile (coalesced 16B loads)
    *reinterpret_cast<float4*>(&As[arow][acol]) =
        *reinterpret_cast<const float4*>(A + (size_t)(m0 + arow) * K + kk + acol);
    // stage B tile transposed into Bst[n][k]
#pragma unroll
    for (int half = 0; half < 2; ++half) {
      const int br = brow + half * 16;
      float4 raw = *reinterpret_cast<const float4*>(
          B + (size_t)(kk + br) * N + n0 + bcol);
      const __bf16* rp = reinterpret_cast<const __bf16*>(&raw);
#pragma unroll
      for (int j = 0; j < 8; ++j) Bst[bcol + j][br] = rp[j];
    }
    __syncthreads();

    // fragments per CDNA5 16-bit A(16x32)/B(32x16) lane layouts
    v16bf afr[2], bfr[2];
#pragma unroll
    for (int mi = 0; mi < 2; ++mi) {
      const int rr = wm * 32 + mi * 16 + l16;
      const int ks = hi ? 8 : 0;       // lanes 0-15: K 0..7 & 16..23; hi: 8..15 & 24..31
      float4* p = reinterpret_cast<float4*>(&afr[mi]);
      p[0] = *reinterpret_cast<const float4*>(&As[rr][ks]);
      p[1] = *reinterpret_cast<const float4*>(&As[rr][ks + 16]);
    }
#pragma unroll
    for (int ni = 0; ni < 2; ++ni) {
      const int cc = wn * 32 + ni * 16 + l16;
      const int ks = hi ? 8 : 0;       // lanes 0-15: K 0..7 & 16..23; hi: 8..15 & 24..31
      float4* p = reinterpret_cast<float4*>(&bfr[ni]);
      p[0] = *reinterpret_cast<const float4*>(&Bst[cc][ks]);
      p[1] = *reinterpret_cast<const float4*>(&Bst[cc][ks + 16]);
    }
#pragma unroll
    for (int mi = 0; mi < 2; ++mi)
#pragma unroll
      for (int ni = 0; ni < 2; ++ni)
        acc[mi][ni] = __builtin_amdgcn_wmma_f32_16x16x32_bf16(
            false, afr[mi], false, bfr[ni], (short)0, acc[mi][ni], false, false);
    __syncthreads();
  }

  // epilogue: C 16x16 layout = VGPR r -> row r (lo) / 8+r (hi), lane -> col
#pragma unroll
  for (int mi = 0; mi < 2; ++mi)
#pragma unroll
    for (int ni = 0; ni < 2; ++ni) {
      const int rowg = m0 + wm * 32 + mi * 16 + (hi ? 8 : 0);
      const int n    = n0 + wn * 32 + ni * 16 + l16;
      const float bb = bias[n];
      float* cp = C + (size_t)rowg * N + n;
#pragma unroll
      for (int r = 0; r < 8; ++r) {
        *cp = acc[mi][ni][r] + bb;
        cp += N;
      }
    }
}

// --------------------------------------------------------------- entmax math
struct PSq  { __device__ float operator()(float u, float) const { return u * u; } };
struct PPow { __device__ float operator()(float u, float ie) const { return __powf(u, ie); } };

// Row-sum of p(Xa - tau) for the 16 register-resident rows of one wave.
// Each lane holds 8 rows x 32 column-tiles; butterfly-reduce over 16 lanes.
template <class PF>
__device__ __forceinline__ void rowsum16(const v8f (&acc)[32],
                                         const float (&tau)[8],
                                         float inv_exp, float (&out)[8]) {
  PF pf;
  float part[8];
#pragma unroll
  for (int r = 0; r < 8; ++r) part[r] = 0.f;
#pragma unroll
  for (int jt = 0; jt < 32; ++jt)
#pragma unroll
    for (int r = 0; r < 8; ++r) {
      float u = fmaxf(acc[jt][r] - tau[r], 0.f);
      part[r] += pf(u, inv_exp);
    }
#pragma unroll
  for (int r = 0; r < 8; ++r) {
    float s = part[r];
#pragma unroll
    for (int sh = 1; sh < 16; sh <<= 1) s += __shfl_xor(s, sh, 32);
    out[r] = s;
  }
}

// Bisection (50 iters) + p@v^T accumulation + bf16 store of res for one
// 16-row tile, entirely from the WMMA C-layout registers.
template <class PF>
__device__ __forceinline__ void process_rows(
    const v8f (&acc)[32], float inv_exp, float hb,
    const float (*vsl)[HDIM], __bf16* __restrict__ resb,
    int t, int rowtile, int l16, bool hi) {
  PF pf;
  float mx[8];
#pragma unroll
  for (int r = 0; r < 8; ++r) mx[r] = -3.0e38f;
#pragma unroll
  for (int jt = 0; jt < 32; ++jt)
#pragma unroll
    for (int r = 0; r < 8; ++r) mx[r] = fmaxf(mx[r], acc[jt][r]);
#pragma unroll
  for (int r = 0; r < 8; ++r)
#pragma unroll
    for (int sh = 1; sh < 16; sh <<= 1)
      mx[r] = fmaxf(mx[r], __shfl_xor(mx[r], sh, 32));

  float tau_lo[8], tau_m[8], f_lo[8];
#pragma unroll
  for (int r = 0; r < 8; ++r) tau_lo[r] = mx[r] - 1.f;
  rowsum16<PF>(acc, tau_lo, inv_exp, f_lo);
#pragma unroll
  for (int r = 0; r < 8; ++r) f_lo[r] -= 1.f;

  float dm = 1.f - hb;  // tau_hi - tau_lo is row-independent
#pragma unroll
  for (int r = 0; r < 8; ++r) tau_m[r] = tau_lo[r];

#pragma unroll 1
  for (int it = 0; it < 50; ++it) {
    dm *= 0.5f;
#pragma unroll
    for (int r = 0; r < 8; ++r) tau_m[r] = tau_lo[r] + dm;
    float fm[8];
    rowsum16<PF>(acc, tau_m, inv_exp, fm);
#pragma unroll
    for (int r = 0; r < 8; ++r)
      if ((fm[r] - 1.f) * f_lo[r] >= 0.f) tau_lo[r] = tau_m[r];
  }

  float S[8], invS[8];
  rowsum16<PF>(acc, tau_m, inv_exp, S);
#pragma unroll
  for (int r = 0; r < 8; ++r) invS[r] = 1.f / S[r];

  // res[h, i] = sum_j p[i, j] * v[h, j]
  float racc[64];
#pragma unroll
  for (int i = 0; i < 64; ++i) racc[i] = 0.f;
#pragma unroll
  for (int jt = 0; jt < 32; ++jt) {
    const int col = jt * 16 + l16;
    float4 v0 = *reinterpret_cast<const float4*>(&vsl[col][0]);
    float4 v1 = *reinterpret_cast<const float4*>(&vsl[col][4]);
    const float cv[8] = {v0.x, v0.y, v0.z, v0.w, v1.x, v1.y, v1.z, v1.w};
    float pv[8];
#pragma unroll
    for (int r = 0; r < 8; ++r) {
      float u = fmaxf(acc[jt][r] - tau_m[r], 0.f);
      pv[r] = pf(u, inv_exp) * invS[r];
    }
#pragma unroll
    for (int h = 0; h < 8; ++h)
#pragma unroll
      for (int r = 0; r < 8; ++r)
        racc[h * 8 + r] = fmaf(cv[h], pv[r], racc[h * 8 + r]);
  }
#pragma unroll
  for (int i = 0; i < 64; ++i)
#pragma unroll
    for (int sh = 1; sh < 16; sh <<= 1)
      racc[i] += __shfl_xor(racc[i], sh, 32);

  const int rbase = rowtile * 16 + (hi ? 8 : 0);
#pragma unroll
  for (int r = 0; r < 8; ++r) {
    if (l16 == r) {
      const int row = rbase + r;
#pragma unroll
      for (int h = 0; h < 8; ++h)
        resb[(size_t)t * EHD + h * EDIM + row] = (__bf16)racc[h * 8 + r];
    }
  }
}

// ------------------------------------------- fused qk^T / entmax / p v^T
// grid = TOK blocks (one token each), 256 threads (8 waves).
// Each wave owns 4 row-tiles of 16 rows; Xa row-block lives in 256 VGPRs.
// rti loop is kept rolled (unroll 1): exactly one 256-VGPR acc set live,
// and the hot bisection loop stays I$-resident.
__global__ __launch_bounds__(256) void fused_entmax_attn(
    const float* __restrict__ qf, const float* __restrict__ kf,
    const float* __restrict__ vf, const float* __restrict__ alphap,
    __bf16* __restrict__ resb) {
  __shared__ __align__(16) __bf16 qs[EDIM][HDIM];  // q^T [i][h]
  __shared__ __align__(16) __bf16 ksd[EDIM][HDIM]; // k^T [j][h]
  __shared__ __align__(16) float  vs[EDIM][HDIM];  // v^T [j][h]

  const int t    = blockIdx.x;
  const int tid  = threadIdx.x;
  const int wave = tid >> 5;
  const int lane = tid & 31;
  const int l16  = lane & 15;
  const bool hi  = lane >= 16;

  const float alpha   = alphap[0];
  const float am1     = alpha - 1.f;
  const float inv_exp = 1.f / am1;
  const bool  sq      = fabsf(inv_exp - 2.f) < 1e-6f;  // alpha == 1.5 fast path
  const float scale   = am1 * rsqrtf((float)EDIM);     // Xa = (alpha-1)*qk/sqrt(E)
  const float hb      = exp2f(-9.f * am1);             // (1/512)^(alpha-1)

  for (int idx = tid; idx < EHD; idx += 256) {
    const int h = idx >> 9;
    const int e = idx & (EDIM - 1);
    qs[e][h]  = (__bf16)qf[(size_t)t * EHD + idx];
    ksd[e][h] = (__bf16)kf[(size_t)t * EHD + idx];
    vs[e][h]  = vf[(size_t)t * EHD + idx];
  }
  __syncthreads();

#pragma unroll 1
  for (int rti = 0; rti < 4; ++rti) {
    const int rt = wave + rti * 8;  // row tile index 0..31

    // A fragment: 16x32 bf16, real K = 8 heads, rest zero-padded
    v16bf afr;
    {
      float4* p = reinterpret_cast<float4*>(&afr);
      p[0] = make_float4(0.f, 0.f, 0.f, 0.f);
      p[1] = make_float4(0.f, 0.f, 0.f, 0.f);
      if (!hi) {
        const int row = rt * 16 + l16;
        p[0] = *reinterpret_cast<const float4*>(&qs[row][0]);  // K=0..7
      }
    }

    v8f acc[32];
#pragma unroll
    for (int jt = 0; jt < 32; ++jt) {
      v16bf bfr;
      float4* p = reinterpret_cast<float4*>(&bfr);
      p[0] = make_float4(0.f, 0.f, 0.f, 0.f);
      p[1] = make_float4(0.f, 0.f, 0.f, 0.f);
      if (!hi) {
        const int col = jt * 16 + l16;
        p[0] = *reinterpret_cast<const float4*>(&ksd[col][0]); // K=0..7
      }
      v8f cz = {0.f, 0.f, 0.f, 0.f, 0.f, 0.f, 0.f, 0.f};
      acc[jt] = __builtin_amdgcn_wmma_f32_16x16x32_bf16(
          false, afr, false, bfr, (short)0, cz, false, false);
    }
#pragma unroll
    for (int jt = 0; jt < 32; ++jt)
#pragma unroll
      for (int r = 0; r < 8; ++r) acc[jt][r] *= scale;

    if (sq) process_rows<PSq >(acc, inv_exp, hb, vs, resb, t, rt, l16, hi);
    else    process_rows<PPow>(acc, inv_exp, hb, vs, resb, t, rt, l16, hi);
  }
}

// ------------------------------------------------------------------ launcher
extern "C" void kernel_launch(void* const* d_in, const int* in_sizes, int n_in,
                              void* d_out, int out_size, void* d_ws, size_t ws_size,
                              hipStream_t stream) {
  (void)in_sizes; (void)n_in; (void)out_size; (void)ws_size;

  const float* x  = (const float*)d_in[0];
  const float* Wq = (const float*)d_in[1];
  const float* bq = (const float*)d_in[2];
  const float* Wk = (const float*)d_in[3];
  const float* bk = (const float*)d_in[4];
  const float* Wv = (const float*)d_in[5];
  const float* bv = (const float*)d_in[6];
  const float* Wu = (const float*)d_in[7];
  const float* bu = (const float*)d_in[8];
  const float* al = (const float*)d_in[9];

  char* wsp = (char*)d_ws;
  size_t off = 0;
  auto alloc = [&](size_t bytes) -> void* {
    void* p = wsp + off;
    off += (bytes + 255) & ~(size_t)255;
    return p;
  };
  __bf16* xb   = (__bf16*)alloc((size_t)TOK * EDIM * 2);
  __bf16* Wqb  = (__bf16*)alloc((size_t)EDIM * EHD * 2);
  __bf16* Wkb  = (__bf16*)alloc((size_t)EDIM * EHD * 2);
  __bf16* Wvb  = (__bf16*)alloc((size_t)EDIM * EHD * 2);
  __bf16* Wub  = (__bf16*)alloc((size_t)EHD * EDIM * 2);
  float*  qfb  = (float*) alloc((size_t)TOK * EHD * 4);
  float*  kfb  = (float*) alloc((size_t)TOK * EHD * 4);
  float*  vfb  = (float*) alloc((size_t)TOK * EHD * 4);
  __bf16* resb = (__bf16*)alloc((size_t)TOK * EHD * 2);

  auto cvt = [&](const float* src, __bf16* dst, int n) {
    f2bf_kernel<<<(n + 255) / 256, 256, 0, stream>>>(src, dst, n);
  };
  cvt(x,  xb,  TOK * EDIM);
  cvt(Wq, Wqb, EDIM * EHD);
  cvt(Wk, Wkb, EDIM * EHD);
  cvt(Wv, Wvb, EDIM * EHD);
  cvt(Wu, Wub, EHD * EDIM);

  dim3 blk(256);
  dim3 gqkv(EHD / 128, TOK / 64);
  gemm_bias_bf16<<<gqkv, blk, 0, stream>>>(xb, Wqb, bq, qfb, TOK, EHD, EDIM);
  gemm_bias_bf16<<<gqkv, blk, 0, stream>>>(xb, Wkb, bk, kfb, TOK, EHD, EDIM);
  gemm_bias_bf16<<<gqkv, blk, 0, stream>>>(xb, Wvb, bv, vfb, TOK, EHD, EDIM);

  fused_entmax_attn<<<TOK, blk, 0, stream>>>(qfb, kfb, vfb, al, resb);

  dim3 gout(EDIM / 128, TOK / 64);
  gemm_bias_bf16<<<gout, blk, 0, stream>>>(resb, Wub, bu, (float*)d_out,
                                           TOK, EDIM, EHD);
}
